// SelfAttention_12292196401662
// MI455X (gfx1250) — matrix operand order repaired and checked by reading, not compiled, not run on hardware
//
#include <hip/hip_runtime.h>

// Self-attention (non-local block), B=4, C=64, N=4096, Cr=8, for gfx1250.
// Flash attention over m with v_wmma_f32_16x16x32_f16. Scores computed
// TRANSPOSED (St[m,n]) so the softmax axis is VGPR-index/half-wave:
//  - Q pre-scaled by log2(e): exp == raw v_exp_f32 (exp2), no per-element mul
//  - row-sum of P computed on the MATRIX pipe via an all-ones A-fragment WMMA
//  - 64 m per iteration: amortizes max-tree and O-rescale
//  - half-wave swap via ds_swizzle SWAPX16 (no index VGPR, no LDS memory)

typedef __attribute__((ext_vector_type(16))) _Float16 v16h;
typedef __attribute__((ext_vector_type(2)))  _Float16 v2h;
typedef __attribute__((ext_vector_type(8)))  float    v8f;

#define BB 4
#define CC 64
#define NN 4096
#define CR 8
#define LOG2E 1.4426950408889634f

static __device__ __forceinline__ float swap16(float v) {
    // ds_swizzle group-of-32: xor=0x10, or=0, and=0x1f -> swap half-waves
    return __int_as_float(__builtin_amdgcn_ds_swizzle(__float_as_int(v), 0x401f));
}

static __device__ __forceinline__ v2h pack2(float v0, float v1) {
    return __builtin_bit_cast(v2h, __builtin_amdgcn_cvt_pkrtz(v0, v1)); // v_cvt_pk_rtz_f16_f32
}

// ---------------------------------------------------------------------------
// Projection q/k -> pre-swizzled WMMA fragments (cr padded 8..31 with zeros).
//   Kf: A-fragment of K^T per 16-m tile   Qf: B-fragment of Q^T per 16-n tile
// Q is pre-multiplied by log2(e) so logits are in the log2 domain.
// ---------------------------------------------------------------------------
__global__ __launch_bounds__(256) void proj_qk_kernel(
    const float* __restrict__ x,
    const float* __restrict__ wq, const float* __restrict__ bq,
    const float* __restrict__ wk, const float* __restrict__ bk,
    _Float16* __restrict__ qf, _Float16* __restrict__ kf)
{
    int gid = blockIdx.x * 256 + threadIdx.x;      // 0 .. B*N-1
    int b = gid >> 12;
    int n = gid & (NN - 1);

    float q[CR], k[CR];
#pragma unroll
    for (int o = 0; o < CR; ++o) { q[o] = bq[o]; k[o] = bk[o]; }

    const float* xp = x + (size_t)b * CC * NN + n;
    for (int c = 0; c < CC; ++c) {
        float xv = xp[(size_t)c * NN];             // coalesced across threads
#pragma unroll
        for (int o = 0; o < CR; ++o) {
            q[o] += wq[o * CC + c] * xv;           // uniform -> s_loads
            k[o] += wk[o * CC + c] * xv;
        }
    }

    int tile = (b << 8) | (n >> 4);
    int l    = n & 15;

    v16h qlo, klo, zer;
#pragma unroll
    for (int h = 0; h < 16; ++h) {
        qlo[h] = (h < CR) ? (_Float16)(q[h] * LOG2E) : (_Float16)0.0f;
        klo[h] = (h < CR) ? (_Float16)k[h]           : (_Float16)0.0f;
        zer[h] = (_Float16)0.0f;
    }
    *(v16h*)(qf + ((size_t)tile * 32 + l)      * 16) = qlo;
    *(v16h*)(qf + ((size_t)tile * 32 + l + 16) * 16) = zer;
    *(v16h*)(kf + ((size_t)tile * 32 + l)      * 16) = klo;
    *(v16h*)(kf + ((size_t)tile * 32 + l + 16) * 16) = zer;
}

// ---------------------------------------------------------------------------
// Projection v -> A-fragments of V (16c x 32m) per (b, c-group, m-chunk).
// 16-bit A layout (contiguous, matches build_bp): lane<16: halves0..15->K0..15;
//                  lane>=16: halves0..15->K16..31.
// ---------------------------------------------------------------------------
__global__ __launch_bounds__(256) void proj_v_kernel(
    const float* __restrict__ x,
    const float* __restrict__ wv, const float* __restrict__ bv,
    _Float16* __restrict__ vf)
{
    int gid = blockIdx.x * 256 + threadIdx.x;
    int b = gid >> 12;
    int m = gid & (NN - 1);

    float v[CC];
#pragma unroll
    for (int co = 0; co < CC; ++co) v[co] = bv[co];

    const float* xp = x + (size_t)b * CC * NN + m;
    for (int ci = 0; ci < CC; ++ci) {
        float xv = xp[(size_t)ci * NN];
#pragma unroll
        for (int co = 0; co < CC; ++co)
            v[co] += wv[co * CC + ci] * xv;
    }

    int mc = m >> 5, mm = m & 31;
    int laneoff = mm & 16;                         // 0 or 16
    int h = mm & 15;
#pragma unroll
    for (int co = 0; co < CC; ++co) {
        int g = co >> 4, cl = co & 15;
        size_t idx = ((((size_t)b * 4 + g) * 128 + mc) * 32 + (cl + laneoff)) * 16 + h;
        vf[idx] = (_Float16)v[co];
    }
}

// ---------------------------------------------------------------------------
// Build B-fragment of P^T (32m x 16n) from two St C/D-layout tiles (exp'd).
// ---------------------------------------------------------------------------
static __device__ __forceinline__ v16h build_bp(const float* p0, const float* p1, bool lo) {
    v16h bp;
#pragma unroll
    for (int j = 0; j < 8; ++j) {
        float v0, v1;
        if (j < 4) {
            float a0 = p0[2 * j], a1 = p0[2 * j + 1];
            float c0 = swap16(p1[2 * j]), c1 = swap16(p1[2 * j + 1]);
            v0 = lo ? a0 : c0;  v1 = lo ? a1 : c1;
        } else {
            int i0 = 2 * j - 8;
            float a0 = swap16(p0[i0]), a1 = swap16(p0[i0 + 1]);
            float c0 = p1[i0], c1 = p1[i0 + 1];
            v0 = lo ? a0 : c0;  v1 = lo ? a1 : c1;
        }
        v2h pk = pack2(v0, v1);
        bp[2 * j]     = pk[0];
        bp[2 * j + 1] = pk[1];
    }
    return bp;
}

// ---------------------------------------------------------------------------
// Flash attention: one wave per 16-query tile; 64 iterations x 64 m.
// Per iter: 4 S-WMMA + 2 sum-WMMA (ones x P^T) + 8 O-WMMA = 14 WMMAs.
// ---------------------------------------------------------------------------
__global__ __launch_bounds__(256) void attn_kernel(
    const float* __restrict__ x, const float* __restrict__ gamma,
    const _Float16* __restrict__ qf, const _Float16* __restrict__ kf,
    const _Float16* __restrict__ vf, float* __restrict__ out)
{
    int lane = threadIdx.x & 31;
    int w    = threadIdx.x >> 5;
    int tile = blockIdx.x * 8 + w;                 // 0..1023
    int b    = tile >> 8;
    int t    = tile & 255;
    int n0   = t << 4;
    bool lo  = lane < 16;

    v16h bqv = *(const v16h*)(qf + ((size_t)tile * 32 + lane) * 16);

    v16h aones;                                    // all-ones A for column sums
#pragma unroll
    for (int h = 0; h < 16; ++h) aones[h] = (_Float16)1.0f;

    v8f ot[4];
    v8f zacc = {};
#pragma unroll
    for (int g = 0; g < 4; ++g) ot[g] = zacc;
    float mprev = -3.0e38f, lsum = 0.0f;

    for (int mc = 0; mc < 64; ++mc) {              // 64 m per iteration
        const _Float16* kbase =
            kf + (((size_t)b * 256 + (size_t)mc * 4) * 32 + lane) * 16;
        v16h ak0 = *(const v16h*)(kbase);
        v16h ak1 = *(const v16h*)(kbase + 512);
        v16h ak2 = *(const v16h*)(kbase + 1024);
        v16h ak3 = *(const v16h*)(kbase + 1536);

        v8f zc = {};
        v8f st0 = __builtin_amdgcn_wmma_f32_16x16x32_f16(false, ak0, false, bqv, (short)0, zc, false, false);
        v8f st1 = __builtin_amdgcn_wmma_f32_16x16x32_f16(false, ak1, false, bqv, (short)0, zc, false, false);
        v8f st2 = __builtin_amdgcn_wmma_f32_16x16x32_f16(false, ak2, false, bqv, (short)0, zc, false, false);
        v8f st3 = __builtin_amdgcn_wmma_f32_16x16x32_f16(false, ak3, false, bqv, (short)0, zc, false, false);

        // ---- column max over 64 m (log2-domain logits) ----
        float cmax = fmaxf(fmaxf(st0[0], st1[0]), fmaxf(st2[0], st3[0]));
#pragma unroll
        for (int r = 1; r < 8; ++r) {
            cmax = fmaxf(cmax, fmaxf(st0[r], st1[r]));
            cmax = fmaxf(cmax, fmaxf(st2[r], st3[r]));
        }
        cmax = fmaxf(cmax, swap16(cmax));
        float mnew  = fmaxf(mprev, cmax);
        float scale = __builtin_amdgcn_exp2f(mprev - mnew);
        mprev = mnew;

        float p0[8], p1[8], p2[8], p3[8];
#pragma unroll
        for (int r = 0; r < 8; ++r) {
            p0[r] = __builtin_amdgcn_exp2f(st0[r] - mnew);
            p1[r] = __builtin_amdgcn_exp2f(st1[r] - mnew);
            p2[r] = __builtin_amdgcn_exp2f(st2[r] - mnew);
            p3[r] = __builtin_amdgcn_exp2f(st3[r] - mnew);
        }

        v16h bp0 = build_bp(p0, p1, lo);           // m 0..31 of chunk
        v16h bp1 = build_bp(p2, p3, lo);           // m 32..63 of chunk

        // ---- row-sum on the matrix pipe: D = ones(16x32) x P^T ----
        v8f dsum = __builtin_amdgcn_wmma_f32_16x16x32_f16(false, aones, false, bp0, (short)0, zc, false, false);
        dsum     = __builtin_amdgcn_wmma_f32_16x16x32_f16(false, aones, false, bp1, (short)0, dsum, false, false);
        lsum = lsum * scale + dsum[0];

        // ---- rescale O accumulators (covers 64 m) ----
#pragma unroll
        for (int g = 0; g < 4; ++g)
#pragma unroll
            for (int r = 0; r < 8; ++r) ot[g][r] *= scale;

        // ---- Ot[g] += V(16c x 32m) x P^T for both 32-m chunks ----
        const _Float16* vbase =
            vf + ((size_t)b * 4 * 128 + (size_t)mc * 2) * 512 + (size_t)lane * 16;
#pragma unroll
        for (int g = 0; g < 4; ++g) {
            v16h av0 = *(const v16h*)(vbase + (size_t)g * 128 * 512);
            v16h av1 = *(const v16h*)(vbase + (size_t)g * 128 * 512 + 512);
            ot[g] = __builtin_amdgcn_wmma_f32_16x16x32_f16(false, av0, false, bp0, (short)0, ot[g], false, false);
            ot[g] = __builtin_amdgcn_wmma_f32_16x16x32_f16(false, av1, false, bp1, (short)0, ot[g], false, false);
        }
    }

    // ---- epilogue: normalize, residual, gamma ----
    float gi = gamma[0] / lsum;
    int nc   = n0 + (lane & 15);
    int coff = (lane >> 4) * 8;
#pragma unroll
    for (int g = 0; g < 4; ++g)
#pragma unroll
        for (int r = 0; r < 8; ++r) {
            int c = g * 16 + r + coff;
            size_t idx = ((size_t)b * CC + c) * NN + nc;
            out[idx] = fmaf(gi, ot[g][r], x[idx]);
        }
}

// ---------------------------------------------------------------------------
extern "C" void kernel_launch(void* const* d_in, const int* in_sizes, int n_in,
                              void* d_out, int out_size, void* d_ws, size_t ws_size,
                              hipStream_t stream) {
    const float* x     = (const float*)d_in[0];
    const float* wq    = (const float*)d_in[1];
    const float* bq    = (const float*)d_in[2];
    const float* wk    = (const float*)d_in[3];
    const float* bk    = (const float*)d_in[4];
    const float* wv    = (const float*)d_in[5];
    const float* bv    = (const float*)d_in[6];
    const float* gamma = (const float*)d_in[7];
    float* out = (float*)d_out;

    // Workspace: Qf 1MB + Kf 1MB + Vf 2MB (f16 fragments)
    _Float16* qf = (_Float16*)d_ws;
    _Float16* kf = qf + (size_t)BB * 256 * 512;
    _Float16* vf = kf + (size_t)BB * 256 * 512;

    proj_qk_kernel<<<64, 256, 0, stream>>>(x, wq, bq, wk, bk, qf, kf);
    proj_v_kernel <<<64, 256, 0, stream>>>(x, wv, bv, vf);
    attn_kernel   <<<128, 256, 0, stream>>>(x, gamma, qf, kf, vf, out);
}
